// Local3DLUT_23278722744853
// MI455X (gfx1250) — matrix operand order repaired and checked by reading, not compiled, not run on hardware
//
#include <hip/hip_runtime.h>
#include <hip/hip_bf16.h>
#include <math.h>

// MI455X / gfx1250 fused Local-3D-LUT kernel.
// Design: memory-bound problem (512MB ctx read dominates; ~23us floor at 23.3TB/s).
// All three 1x1-conv GEMMs run on v_wmma_f32_16x16x32_f16 (f16 inputs, f32 acc),
// fully fused so h/h2 never touch HBM. 16 pixels = one WMMA N-tile per wave.

typedef __attribute__((ext_vector_type(16))) _Float16 v16h;
typedef __attribute__((ext_vector_type(8)))  _Float16 v8h;
typedef __attribute__((ext_vector_type(4)))  _Float16 v4h;
typedef __attribute__((ext_vector_type(8)))  float    v8f;

#define CTX    128
#define HW     262144          // 512*512
#define NPIX   1048576         // 4*512*512
#define ROWH   132             // LDS halfs per pixel row (264B: 8B aligned, stride 66 dwords)

static __device__ __forceinline__ v8h cat44(v4h a, v4h b) {
  return __builtin_shufflevector(a, b, 0,1,2,3,4,5,6,7);
}
static __device__ __forceinline__ v16h cat88(v8h a, v8h b) {
  return __builtin_shufflevector(a, b, 0,1,2,3,4,5,6,7,8,9,10,11,12,13,14,15);
}

static __device__ __forceinline__ float gelu_exact(float x) {
  return 0.5f * x * (1.0f + erff(x * 0.70710678118654752f));
}

// ---------------- weight conversion (f32 -> f16, W3 padded to 16x64) ----------------
__global__ void prep_weights_kernel(const float* __restrict__ W1,
                                    const float* __restrict__ W2,
                                    const float* __restrict__ W3,
                                    _Float16* __restrict__ W1h,
                                    _Float16* __restrict__ W2h,
                                    _Float16* __restrict__ W3h) {
  int i = blockIdx.x * 256 + threadIdx.x;
  if (i < 128 * 128) W1h[i] = (_Float16)W1[i];
  if (i < 64 * 128)  W2h[i] = (_Float16)W2[i];
  if (i < 16 * 64) {
    int r = i >> 6, c = i & 63;
    W3h[i] = (_Float16)((r < 3) ? W3[r * 64 + c] : 0.0f);
  }
}

// ---------------- fused MLP + trilinear LUT ----------------
__global__ __launch_bounds__(256)
void local3dlut_fused_kernel(const float* __restrict__ image,
                             const float* __restrict__ ctx,
                             const float* __restrict__ lut,     // [33][33][33][3]
                             const float* __restrict__ b1,
                             const float* __restrict__ b2,
                             const float* __restrict__ b3,
                             const float* __restrict__ osc_p,
                             const _Float16* __restrict__ W1h,  // [128][128]
                             const _Float16* __restrict__ W2h,  // [64][128]
                             const _Float16* __restrict__ W3h,  // [16][64]
                             float* __restrict__ out) {
  __shared__ _Float16 sBuf[128 * ROWH];   // [pixel 0..127][channel], 33,792 B

  const int t    = threadIdx.x;
  const int lane = t & 31;
  const int wv   = t >> 5;          // wave id 0..7 -> 16-pixel tile
  const int g    = lane >> 4;       // half-wave select
  const int l16  = lane & 15;

  const int pBase = blockIdx.x * 128;          // flat pixel base over B*H*W
  const int bIdx  = pBase / HW;
  const int s0    = pBase % HW;                // offset inside one image plane
  const float* ctxB = ctx + (size_t)bIdx * CTX * HW + s0;

  // ---- cooperative stage: ctx[128ch x 128px] f32 -> f16 LDS [px][ch] ----
  #pragma unroll
  for (int it = 0; it < 16; ++it) {
    int e  = it * 256 + t;             // 4096 float4 units
    int ch = e >> 5;                   // 32 float4 per channel row
    int p4 = (e & 31) << 2;
    float4 v = *(const float4*)(ctxB + (size_t)ch * HW + p4);
    sBuf[(p4 + 0) * ROWH + ch] = (_Float16)v.x;
    sBuf[(p4 + 1) * ROWH + ch] = (_Float16)v.y;
    sBuf[(p4 + 2) * ROWH + ch] = (_Float16)v.z;
    sBuf[(p4 + 3) * ROWH + ch] = (_Float16)v.w;
  }
  __syncthreads();

  _Float16* rowp = &sBuf[(wv * 16 + l16) * ROWH];

  // B fragment (32x16 f16): lane = column l16, contiguous 16 K values at k0 + 16*g
  auto ldB = [&](int k0) -> v16h {
    const _Float16* p = rowp + k0 + 16 * g;
    v4h a = *(const v4h*)(p + 0);
    v4h b = *(const v4h*)(p + 4);
    v4h c = *(const v4h*)(p + 8);
    v4h d = *(const v4h*)(p + 12);
    return cat88(cat44(a, b), cat44(c, d));
  };
  // A fragment (16x32 f16): row striped in-lane, contiguous 16 K values at k0 + 16*g
  auto ldA = [&](const _Float16* W, int ldk, int row, int k0) -> v16h {
    const _Float16* p = W + row * ldk + k0 + 16 * g;
    v8h lo = *(const v8h*)(p);        // K = k0 + 16g .. +7
    v8h hi = *(const v8h*)(p + 8);    // K = k0 + 16g + 8 .. +7
    return cat88(lo, hi);
  };

  // ================= GEMM1: h[128x16] = gelu(W1 @ ctx + b1) =================
  v16h bf0 = ldB(0), bf1 = ldB(32), bf2 = ldB(64), bf3 = ldB(96);

  v8f acc[8];
  #pragma unroll
  for (int mt = 0; mt < 8; ++mt) {
    const float4* bp = (const float4*)(b1 + mt * 16 + 8 * g);
    float4 x = bp[0], y = bp[1];
    acc[mt][0] = x.x; acc[mt][1] = x.y; acc[mt][2] = x.z; acc[mt][3] = x.w;
    acc[mt][4] = y.x; acc[mt][5] = y.y; acc[mt][6] = y.z; acc[mt][7] = y.w;
  }
  #pragma unroll
  for (int mt = 0; mt < 8; ++mt) {
    int row = mt * 16 + l16;
    acc[mt] = __builtin_amdgcn_wmma_f32_16x16x32_f16(false, ldA(W1h, 128, row, 0),  false, bf0, (short)0, acc[mt], false, false);
    acc[mt] = __builtin_amdgcn_wmma_f32_16x16x32_f16(false, ldA(W1h, 128, row, 32), false, bf1, (short)0, acc[mt], false, false);
    acc[mt] = __builtin_amdgcn_wmma_f32_16x16x32_f16(false, ldA(W1h, 128, row, 64), false, bf2, (short)0, acc[mt], false, false);
    acc[mt] = __builtin_amdgcn_wmma_f32_16x16x32_f16(false, ldA(W1h, 128, row, 96), false, bf3, (short)0, acc[mt], false, false);
  }
  // GELU + write h back into this wave's LDS rows (frags already in registers)
  #pragma unroll
  for (int mt = 0; mt < 8; ++mt) {
    #pragma unroll
    for (int r = 0; r < 8; ++r) {
      rowp[mt * 16 + 8 * g + r] = (_Float16)gelu_exact(acc[mt][r]);
    }
  }

  // ================= GEMM2: h2[64x16] = gelu(W2 @ h + b2) =================
  v16h cf0 = ldB(0), cf1 = ldB(32), cf2 = ldB(64), cf3 = ldB(96);
  v8f acc2[4];
  #pragma unroll
  for (int mt = 0; mt < 4; ++mt) {
    const float4* bp = (const float4*)(b2 + mt * 16 + 8 * g);
    float4 x = bp[0], y = bp[1];
    acc2[mt][0] = x.x; acc2[mt][1] = x.y; acc2[mt][2] = x.z; acc2[mt][3] = x.w;
    acc2[mt][4] = y.x; acc2[mt][5] = y.y; acc2[mt][6] = y.z; acc2[mt][7] = y.w;
  }
  #pragma unroll
  for (int mt = 0; mt < 4; ++mt) {
    int row = mt * 16 + l16;
    acc2[mt] = __builtin_amdgcn_wmma_f32_16x16x32_f16(false, ldA(W2h, 128, row, 0),  false, cf0, (short)0, acc2[mt], false, false);
    acc2[mt] = __builtin_amdgcn_wmma_f32_16x16x32_f16(false, ldA(W2h, 128, row, 32), false, cf1, (short)0, acc2[mt], false, false);
    acc2[mt] = __builtin_amdgcn_wmma_f32_16x16x32_f16(false, ldA(W2h, 128, row, 64), false, cf2, (short)0, acc2[mt], false, false);
    acc2[mt] = __builtin_amdgcn_wmma_f32_16x16x32_f16(false, ldA(W2h, 128, row, 96), false, cf3, (short)0, acc2[mt], false, false);
  }
  #pragma unroll
  for (int mt = 0; mt < 4; ++mt) {
    #pragma unroll
    for (int r = 0; r < 8; ++r) {
      rowp[mt * 16 + 8 * g + r] = (_Float16)gelu_exact(acc2[mt][r]);
    }
  }

  // ================= GEMM3: off[3x16] = tanh(W3 @ h2 + b3) * scale =================
  v16h df0 = ldB(0), df1 = ldB(32);
  v8f acc3;
  acc3[0] = g ? 0.0f : b3[0];
  acc3[1] = g ? 0.0f : b3[1];
  acc3[2] = g ? 0.0f : b3[2];
  acc3[3] = 0.0f; acc3[4] = 0.0f; acc3[5] = 0.0f; acc3[6] = 0.0f; acc3[7] = 0.0f;
  acc3 = __builtin_amdgcn_wmma_f32_16x16x32_f16(false, ldA(W3h, 64, l16, 0),  false, df0, (short)0, acc3, false, false);
  acc3 = __builtin_amdgcn_wmma_f32_16x16x32_f16(false, ldA(W3h, 64, l16, 32), false, df1, (short)0, acc3, false, false);

  // ================= epilogue: LUT sample + offset (after all WMMAs) =================
  const float osc = *osc_p;
  if (g == 0) {                      // lanes 0..15 hold rows 0..2 of the offset tile
    float off0 = tanhf(acc3[0]) * osc;
    float off1 = tanhf(acc3[1]) * osc;
    float off2 = tanhf(acc3[2]) * osc;

    int s = s0 + wv * 16 + l16;
    const float* img = image + (size_t)bIdx * 3 * HW + s;
    float R  = img[0];
    float G  = img[HW];
    float Bv = img[2 * HW];

    float r01 = fminf(fmaxf((R  + 1.0f) * 0.5f, 0.0f), 1.0f);
    float g01 = fminf(fmaxf((G  + 1.0f) * 0.5f, 0.0f), 1.0f);
    float b01 = fminf(fmaxf((Bv + 1.0f) * 0.5f, 0.0f), 1.0f);
    float gx = r01 * 32.0f;   // R -> x (last lut axis)
    float gy = g01 * 32.0f;   // G -> y
    float gz = b01 * 32.0f;   // B -> z (first lut axis)

    int x0 = (int)gx, y0 = (int)gy, z0 = (int)gz;   // gx,gy,gz in [0,32]
    int x1 = min(x0 + 1, 32), y1 = min(y0 + 1, 32), z1 = min(z0 + 1, 32);
    float fx = gx - (float)x0, fy = gy - (float)y0, fz = gz - (float)z0;
    float ox = 1.0f - fx, oy = 1.0f - fy, oz = 1.0f - fz;

    const float* c000 = lut + (size_t)(((z0 * 33) + y0) * 33 + x0) * 3;
    const float* c001 = lut + (size_t)(((z0 * 33) + y0) * 33 + x1) * 3;
    const float* c010 = lut + (size_t)(((z0 * 33) + y1) * 33 + x0) * 3;
    const float* c011 = lut + (size_t)(((z0 * 33) + y1) * 33 + x1) * 3;
    const float* c100 = lut + (size_t)(((z1 * 33) + y0) * 33 + x0) * 3;
    const float* c101 = lut + (size_t)(((z1 * 33) + y0) * 33 + x1) * 3;
    const float* c110 = lut + (size_t)(((z1 * 33) + y1) * 33 + x0) * 3;
    const float* c111 = lut + (size_t)(((z1 * 33) + y1) * 33 + x1) * 3;

    float w000 = ox * oy * oz, w001 = fx * oy * oz;
    float w010 = ox * fy * oz, w011 = fx * fy * oz;
    float w100 = ox * oy * fz, w101 = fx * oy * fz;
    float w110 = ox * fy * fz, w111 = fx * fy * fz;

    float offs0[3] = {off0, off1, off2};
    float* outp = out + (size_t)bIdx * 3 * HW + s;
    #pragma unroll
    for (int c = 0; c < 3; ++c) {
      float sv = w000 * c000[c] + w001 * c001[c] + w010 * c010[c] + w011 * c011[c]
               + w100 * c100[c] + w101 * c101[c] + w110 * c110[c] + w111 * c111[c];
      outp[c * HW] = sv * 2.0f - 1.0f + offs0[c];
    }
  }
}

extern "C" void kernel_launch(void* const* d_in, const int* in_sizes, int n_in,
                              void* d_out, int out_size, void* d_ws, size_t ws_size,
                              hipStream_t stream) {
  const float* image = (const float*)d_in[0];
  const float* ctx   = (const float*)d_in[1];
  const float* lut   = (const float*)d_in[2];
  const float* W1    = (const float*)d_in[3];
  const float* b1    = (const float*)d_in[4];
  const float* W2    = (const float*)d_in[5];
  const float* b2    = (const float*)d_in[6];
  const float* W3    = (const float*)d_in[7];
  const float* b3    = (const float*)d_in[8];
  const float* osc   = (const float*)d_in[9];

  _Float16* W1h = (_Float16*)d_ws;          // 128*128 halfs
  _Float16* W2h = W1h + 128 * 128;          // 64*128 halfs
  _Float16* W3h = W2h + 64 * 128;           // 16*64 halfs (padded)

  prep_weights_kernel<<<64, 256, 0, stream>>>(W1, W2, W3, W1h, W2h, W3h);
  local3dlut_fused_kernel<<<NPIX / 128, 256, 0, stream>>>(
      image, ctx, lut, b1, b2, b3, osc, W1h, W2h, W3h, (float*)d_out);
}
